// TransformerModelv16_833223655863
// MI455X (gfx1250) — matrix operand order repaired and checked by reading, not compiled, not run on hardware
//
#include <hip/hip_runtime.h>
#include <stdint.h>
#include <stddef.h>

typedef __attribute__((ext_vector_type(16))) _Float16 v16h;
typedef __attribute__((ext_vector_type(8)))  _Float16 v8h;
typedef __attribute__((ext_vector_type(8)))  float    v8f;

#define EPSX 1e-5f

// =====================================================================
// WMMA GEMM:  C(M,N) = act(A(M,K) * B^T(N,K) + bias)
// A: (Mp, Kp) f16 row-major, Mp = ceil128(M), rows >= M are zero.
// B: (Np, Kp) f16 row-major (pre-transposed weights), pads are zero.
// Kp multiple of 32. Block tile 128x64x32, 256 threads = 8 waves in a
// 4x2 grid; each wave owns a 32x32 macro-tile = 2x2 WMMA tiles.
// All tile loads are unconditional async-to-LDS b128 with double-
// buffered LDS; ASYNCcnt paces the pipeline (3 asyncs per tile-step).
// ACT: 0 none, 1 relu, 2 gelu(exact), 3 sigmoid
// =====================================================================
#define ASYNC_B128(lds, gptr)                                              \
  asm volatile("global_load_async_to_lds_b128 %0, %1, off"                 \
               :: "v"(lds), "v"((uint64_t)(uintptr_t)(gptr)) : "memory")

template<int ACT>
__global__ __launch_bounds__(256)
void k_gemm(const _Float16* __restrict__ A, const _Float16* __restrict__ B,
            const float* __restrict__ bias, float* __restrict__ Cf,
            _Float16* __restrict__ Ch, int M, int N, int Kp)
{
  __shared__ _Float16 As[2][128][32];  // 2 x 8 KB
  __shared__ _Float16 Bs[2][64][32];   // 2 x 4 KB (n-major)

  const int tid  = threadIdx.x;
  const int lane = tid & 31;
  const int wave = tid >> 5;
  const int rowl = lane & 15;
  const int half = lane >> 4;
  const int m0 = blockIdx.y * 128;
  const int n0 = blockIdx.x * 64;
  const int wn = wave & 1;            // n half: col offset wn*32
  const int wm = wave >> 1;           // m quarter: row offset wm*32

  v8f acc[2][2] = {};

  const int tr  = tid >> 2;           // 0..63
  const int tk8 = (tid & 3) << 3;     // 0,8,16,24
  const uint32_t lAlo0 = (uint32_t)(uintptr_t)(&As[0][tr][tk8]);
  const uint32_t lAlo1 = (uint32_t)(uintptr_t)(&As[1][tr][tk8]);
  const uint32_t lAhi0 = (uint32_t)(uintptr_t)(&As[0][tr + 64][tk8]);
  const uint32_t lAhi1 = (uint32_t)(uintptr_t)(&As[1][tr + 64][tk8]);
  const uint32_t lB0   = (uint32_t)(uintptr_t)(&Bs[0][tr][tk8]);
  const uint32_t lB1   = (uint32_t)(uintptr_t)(&Bs[1][tr][tk8]);
  const _Float16* aLo = A + (size_t)(m0 + tr) * Kp + tk8;
  const _Float16* aHi = A + (size_t)(m0 + tr + 64) * Kp + tk8;
  const _Float16* bp  = B + (size_t)(n0 + tr) * Kp + tk8;

  const int nt = Kp >> 5;
  // prologue: tile 0 -> buffer 0
  ASYNC_B128(lAlo0, aLo);
  ASYNC_B128(lAhi0, aHi);
  ASYNC_B128(lB0, bp);

  for (int i = 0; i < nt; ++i) {
    const int cur = i & 1;
    if (i + 1 < nt) {
      // stage tile i+1 into the other buffer while tile i computes
      ASYNC_B128(cur ? lAlo0 : lAlo1, aLo + (size_t)(i + 1) * 32);
      ASYNC_B128(cur ? lAhi0 : lAhi1, aHi + (size_t)(i + 1) * 32);
      ASYNC_B128(cur ? lB0 : lB1, bp + (size_t)(i + 1) * 32);
      asm volatile("s_wait_asynccnt 0x3" ::: "memory");  // tile i landed
    } else {
      asm volatile("s_wait_asynccnt 0x0" ::: "memory");
    }
    __syncthreads();

    // fragments per ISA wave32 layouts
    v16h a0, a1, b0, b1;
    const int r0 = (wm << 5) + rowl;
    ((v8h*)&a0)[0] = *(const v8h*)(&As[cur][r0][half << 3]);
    ((v8h*)&a0)[1] = *(const v8h*)(&As[cur][r0][16 + (half << 3)]);
    ((v8h*)&a1)[0] = *(const v8h*)(&As[cur][r0 + 16][half << 3]);
    ((v8h*)&a1)[1] = *(const v8h*)(&As[cur][r0 + 16][16 + (half << 3)]);
    const int c0 = (wn << 5) + rowl;
    ((v8h*)&b0)[0] = *(const v8h*)(&Bs[cur][c0][half << 3]);
    ((v8h*)&b0)[1] = *(const v8h*)(&Bs[cur][c0][16 + (half << 3)]);
    ((v8h*)&b1)[0] = *(const v8h*)(&Bs[cur][c0 + 16][half << 3]);
    ((v8h*)&b1)[1] = *(const v8h*)(&Bs[cur][c0 + 16][16 + (half << 3)]);

    acc[0][0] = __builtin_amdgcn_wmma_f32_16x16x32_f16(false, a0, false, b0,
                                                       (short)0, acc[0][0], false, false);
    acc[0][1] = __builtin_amdgcn_wmma_f32_16x16x32_f16(false, a0, false, b1,
                                                       (short)0, acc[0][1], false, false);
    acc[1][0] = __builtin_amdgcn_wmma_f32_16x16x32_f16(false, a1, false, b0,
                                                       (short)0, acc[1][0], false, false);
    acc[1][1] = __builtin_amdgcn_wmma_f32_16x16x32_f16(false, a1, false, b1,
                                                       (short)0, acc[1][1], false, false);
    __syncthreads();
  }

  // epilogue: bias + activation, f32/f16 stores (logical bounds only here)
  #pragma unroll
  for (int ni = 0; ni < 2; ++ni) {
    const int cn = n0 + (wn << 5) + (ni << 4) + rowl;
    if (cn < N) {
      const float bv = (bias != nullptr) ? bias[cn] : 0.f;
      #pragma unroll
      for (int mi = 0; mi < 2; ++mi) {
        #pragma unroll
        for (int i = 0; i < 8; ++i) {
          const int cm = m0 + (wm << 5) + (mi << 4) + i + (half << 3);
          if (cm < M) {
            float v = acc[mi][ni][i] + bv;
            if (ACT == 1) v = fmaxf(v, 0.f);
            else if (ACT == 2) v = 0.5f * v * (1.f + erff(v * 0.70710678118f));
            else if (ACT == 3) v = 1.f / (1.f + expf(-v));
            if (Cf) Cf[(size_t)cm * N + cn] = v;
            if (Ch) Ch[(size_t)cm * N + cn] = (_Float16)v;
          }
        }
      }
    }
  }
}

// =====================================================================
// im2col (NHWC f16 in -> col f16, (Mp, Kp) padded with zeros)
// =====================================================================
__global__ void k_im2col(const _Float16* __restrict__ x, _Float16* __restrict__ col,
                         int Nimg, int H, int W, int C, int Ho, int Wo,
                         int KH, int KW, int stride, int pad, int Kp, int Mp)
{
  const size_t total = (size_t)Mp * Kp;
  const size_t Mreal = (size_t)Nimg * Ho * Wo;
  const int K = KH * KW * C;
  for (size_t idx = (size_t)blockIdx.x * blockDim.x + threadIdx.x; idx < total;
       idx += (size_t)gridDim.x * blockDim.x) {
    const int k = (int)(idx % Kp);
    const size_t m = idx / Kp;
    _Float16 v = (_Float16)0.f;
    if (k < K && m < Mreal) {
      const int ci = k % C;
      const int r  = k / C;
      const int kx = r % KW;
      const int ky = r / KW;
      const int ox = (int)(m % Wo);
      const size_t t = m / Wo;
      const int oy = (int)(t % Ho);
      const int n  = (int)(t / Ho);
      const int iy = oy * stride - pad + ky;
      const int ix = ox * stride - pad + kx;
      if (iy >= 0 && iy < H && ix >= 0 && ix < W)
        v = x[(((size_t)n * H + iy) * W + ix) * C + ci];
    }
    col[idx] = v;
  }
}

// deconv (ConvTranspose2d k3 s2 p1 op1) as conv over 2x-dilated input
__global__ void k_im2col_dec(const _Float16* __restrict__ x, _Float16* __restrict__ col,
                             int Nimg, int H, int W, int C, int Ho, int Wo, int Mp)
{
  const int Kp = 9 * C;  // multiple of 32 for all decoder layers
  const size_t total = (size_t)Mp * Kp;
  const size_t Mreal = (size_t)Nimg * Ho * Wo;
  for (size_t idx = (size_t)blockIdx.x * blockDim.x + threadIdx.x; idx < total;
       idx += (size_t)gridDim.x * blockDim.x) {
    const int k = (int)(idx % Kp);
    const size_t m = idx / Kp;
    _Float16 v = (_Float16)0.f;
    if (m < Mreal) {
      const int ci = k % C;
      const int r  = k / C;
      const int kx = r % 3;
      const int ky = r / 3;
      const int ox = (int)(m % Wo);
      const size_t t = m / Wo;
      const int oy = (int)(t % Ho);
      const int n  = (int)(t / Ho);
      const int uy = oy + ky - 1;
      const int ux = ox + kx - 1;
      if (uy >= 0 && ux >= 0 && (uy & 1) == 0 && (ux & 1) == 0) {
        const int iy = uy >> 1, ix = ux >> 1;
        if (iy < H && ix < W)
          v = x[(((size_t)n * H + iy) * W + ix) * C + ci];
      }
    }
    col[idx] = v;
  }
}

// =====================================================================
// weight converters: f32 -> f16, emitted pre-transposed as (Np, Kp)
// =====================================================================
__global__ void k_wconv(const float* __restrict__ w, _Float16* __restrict__ wt,
                        int Co, int Ci, int KH, int KW, int Kp, int Np)
{
  const size_t total = (size_t)Np * Kp;
  const int K = KH * KW * Ci;
  for (size_t idx = (size_t)blockIdx.x * blockDim.x + threadIdx.x; idx < total;
       idx += (size_t)gridDim.x * blockDim.x) {
    const int k = (int)(idx % Kp);
    const int n = (int)(idx / Kp);
    _Float16 v = (_Float16)0.f;
    if (n < Co && k < K) {
      const int ci = k % Ci;
      const int r  = k / Ci;
      const int kx = r % KW;
      const int ky = r / KW;
      v = (_Float16)w[(((size_t)n * Ci + ci) * KH + ky) * KW + kx];
    }
    wt[idx] = v;
  }
}

// deconv weights: torch (Ci,Co,3,3); fold flip + transpose; out (Np, 9*Ci)
__global__ void k_wdeconv(const float* __restrict__ w, _Float16* __restrict__ wt,
                          int Ci, int Co, int Np)
{
  const int Kp = 9 * Ci;
  const size_t total = (size_t)Np * Kp;
  for (size_t idx = (size_t)blockIdx.x * blockDim.x + threadIdx.x; idx < total;
       idx += (size_t)gridDim.x * blockDim.x) {
    const int k = (int)(idx % Kp);
    const int n = (int)(idx / Kp);
    _Float16 v = (_Float16)0.f;
    if (n < Co) {
      const int ci = k % Ci;
      const int r  = k / Ci;
      const int kx = r % 3;
      const int ky = r / 3;
      v = (_Float16)w[(((size_t)ci * Co + n) * 3 + (2 - ky)) * 3 + (2 - kx)];
    }
    wt[idx] = v;
  }
}

// linear weights: w (Dout,Din) f32 -> wt (Dout,Din) f16 (already B^T layout)
// mode 0: straight copy
// mode 1: encoder fc (K index is NHWC (h,w,c), source col is NCHW flat)
// mode 2: decoder fc (N index is NHWC (h,w,c), source row is NCHW flat)
__global__ void k_wlin(const float* __restrict__ w, _Float16* __restrict__ wt,
                       int Dout, int Din, int mode)
{
  const size_t total = (size_t)Dout * Din;
  for (size_t idx = (size_t)blockIdx.x * blockDim.x + threadIdx.x; idx < total;
       idx += (size_t)gridDim.x * blockDim.x) {
    const int k = (int)(idx % Din);
    const int n = (int)(idx / Din);
    float v;
    if (mode == 1) {
      const int c = k % 256, hw = k / 256;
      const int wx = hw % 10, hy = hw / 10;
      const int kk = c * 100 + hy * 10 + wx;
      v = w[(size_t)n * Din + kk];
    } else if (mode == 2) {
      const int c = n % 256, hw = n / 256;
      const int wx = hw % 10, hy = hw / 10;
      const int nn = c * 100 + hy * 10 + wx;
      v = w[(size_t)nn * Din + k];
    } else {
      v = w[idx];
    }
    wt[idx] = (_Float16)v;
  }
}

__global__ void k_decbias(const float* __restrict__ b, float* __restrict__ bp)
{
  for (size_t n = (size_t)blockIdx.x * blockDim.x + threadIdx.x; n < 25600;
       n += (size_t)gridDim.x * blockDim.x) {
    const int c = (int)(n % 256), hw = (int)(n / 256);
    const int wx = hw % 10, hy = hw / 10;
    bp[n] = b[c * 100 + hy * 10 + wx];
  }
}

// =====================================================================
// batchnorm (training-mode batch stats over rows, channels = columns)
// =====================================================================
__global__ void k_zero(float* p, int n)
{
  int i = blockIdx.x * blockDim.x + threadIdx.x;
  if (i < n) p[i] = 0.f;
}

__global__ void k_zero_f16(_Float16* p, size_t n)
{
  for (size_t i = (size_t)blockIdx.x * blockDim.x + threadIdx.x; i < n;
       i += (size_t)gridDim.x * blockDim.x)
    p[i] = (_Float16)0.f;
}

__global__ __launch_bounds__(256)
void k_bnstats(const float* __restrict__ x, float* __restrict__ sum,
               float* __restrict__ sq, long long Mrows, int C)
{
  const int c = blockIdx.y;
  float s = 0.f, s2 = 0.f;
  for (long long m = (long long)blockIdx.x * blockDim.x + threadIdx.x; m < Mrows;
       m += (long long)gridDim.x * blockDim.x) {
    const float v = x[(size_t)m * C + c];
    s += v; s2 += v * v;
  }
  __shared__ float sh0[256], sh1[256];
  sh0[threadIdx.x] = s; sh1[threadIdx.x] = s2;
  __syncthreads();
  for (int o = 128; o > 0; o >>= 1) {
    if ((int)threadIdx.x < o) {
      sh0[threadIdx.x] += sh0[threadIdx.x + o];
      sh1[threadIdx.x] += sh1[threadIdx.x + o];
    }
    __syncthreads();
  }
  if (threadIdx.x == 0) { atomicAdd(&sum[c], sh0[0]); atomicAdd(&sq[c], sh1[0]); }
}

__global__ void k_bn_apply_relu(const float* __restrict__ x, const float* __restrict__ sum,
                                const float* __restrict__ sq, const float* __restrict__ g,
                                const float* __restrict__ b, _Float16* __restrict__ y,
                                size_t Mrows, int C)
{
  const size_t total = Mrows * C;
  const float invM = 1.f / (float)Mrows;
  for (size_t idx = (size_t)blockIdx.x * blockDim.x + threadIdx.x; idx < total;
       idx += (size_t)gridDim.x * blockDim.x) {
    const int c = (int)(idx % C);
    const float m = sum[c] * invM;
    const float var = sq[c] * invM - m * m;
    float v = (x[idx] - m) * rsqrtf(var + EPSX) * g[c] + b[c];
    y[idx] = (_Float16)fmaxf(v, 0.f);
  }
}

// fused: relu(bn_A(a) + bn_B(b)) -> f16
__global__ void k_bn2_add_relu(const float* __restrict__ xa, const float* __restrict__ xb,
                               const float* sumA, const float* sqA,
                               const float* ga, const float* ba,
                               const float* sumB, const float* sqB,
                               const float* gb, const float* bb,
                               _Float16* __restrict__ y, size_t Mrows, int C)
{
  const size_t total = Mrows * C;
  const float invM = 1.f / (float)Mrows;
  for (size_t idx = (size_t)blockIdx.x * blockDim.x + threadIdx.x; idx < total;
       idx += (size_t)gridDim.x * blockDim.x) {
    const int c = (int)(idx % C);
    const float mA = sumA[c] * invM, vA = sqA[c] * invM - mA * mA;
    const float mB = sumB[c] * invM, vB = sqB[c] * invM - mB * mB;
    const float a = (xa[idx] - mA) * rsqrtf(vA + EPSX) * ga[c] + ba[c];
    const float b = (xb[idx] - mB) * rsqrtf(vB + EPSX) * gb[c] + bb[c];
    y[idx] = (_Float16)fmaxf(a + b, 0.f);
  }
}

// =====================================================================
// small/elementwise kernels
// =====================================================================
__global__ void k_f32_to_f16(const float* __restrict__ x, _Float16* __restrict__ y, size_t n)
{
  for (size_t i = (size_t)blockIdx.x * blockDim.x + threadIdx.x; i < n;
       i += (size_t)gridDim.x * blockDim.x)
    y[i] = (_Float16)x[i];
}

__global__ void k_add(const float* __restrict__ a, const float* __restrict__ b,
                      float* __restrict__ c, size_t n)
{
  for (size_t i = (size_t)blockIdx.x * blockDim.x + threadIdx.x; i < n;
       i += (size_t)gridDim.x * blockDim.x)
    c[i] = a[i] + b[i];
}

// one wave32 per row layernorm; optional f32 and f16 outputs
__global__ __launch_bounds__(32)
void k_layernorm(const float* __restrict__ x, const float* __restrict__ g,
                 const float* __restrict__ b, float* __restrict__ yf,
                 _Float16* __restrict__ yh, int D)
{
  const int r = blockIdx.x;
  const int lane = threadIdx.x;
  float s = 0.f, s2 = 0.f;
  for (int d = lane; d < D; d += 32) {
    const float v = x[(size_t)r * D + d];
    s += v; s2 += v * v;
  }
  for (int o = 16; o > 0; o >>= 1) { s += __shfl_xor(s, o, 32); s2 += __shfl_xor(s2, o, 32); }
  const float m = s / D;
  const float inv = rsqrtf(s2 / D - m * m + EPSX);
  for (int d = lane; d < D; d += 32) {
    const float v = (x[(size_t)r * D + d] - m) * inv * g[d] + b[d];
    if (yf) yf[(size_t)r * D + d] = v;
    if (yh) yh[(size_t)r * D + d] = (_Float16)v;
  }
}

// MAE-style 2D sin-cos pos embedding, computed on the fly, added to x (8,9,768)
__global__ void k_posembed(float* __restrict__ x)
{
  const size_t total = (size_t)72 * 768;
  for (size_t idx = (size_t)blockIdx.x * blockDim.x + threadIdx.x; idx < total;
       idx += (size_t)gridDim.x * blockDim.x) {
    const int d = (int)(idx % 768);
    const int p = (int)((idx / 768) % 9);
    const float gw = (float)(p % 3), gh = (float)(p / 3);
    const int q = d % 384;
    const float pos = (d / 384) ? gh : gw;
    const int tt = q % 192;
    const float omega = powf(10000.f, -((float)tt) / 192.f);
    const float ang = pos * omega;
    x[idx] += (q / 192) ? cosf(ang) : sinf(ang);
  }
}

// TemporalContextNorm over the 9 panels: per (b,d)
__global__ void k_tcn_fwd(float* __restrict__ x, float* __restrict__ mbuf,
                          float* __restrict__ vbuf, const float* __restrict__ g,
                          const float* __restrict__ b)
{
  const int idx = blockIdx.x * blockDim.x + threadIdx.x;
  if (idx >= 8 * 768) return;
  const int bb = idx / 768, d = idx % 768;
  float s = 0.f, s2 = 0.f;
  for (int p = 0; p < 9; ++p) {
    const float v = x[((size_t)bb * 9 + p) * 768 + d];
    s += v; s2 += v * v;
  }
  const float m = s / 9.f, var = s2 / 9.f - m * m;
  mbuf[idx] = m; vbuf[idx] = var;
  const float inv = rsqrtf(var + EPSX);
  for (int p = 0; p < 9; ++p) {
    const size_t o = ((size_t)bb * 9 + p) * 768 + d;
    x[o] = (x[o] - m) * inv * g[d] + b[d];
  }
}

__global__ void k_tcn_inv(float* __restrict__ x, const float* __restrict__ mbuf,
                          const float* __restrict__ vbuf, const float* __restrict__ g,
                          const float* __restrict__ b)
{
  const int idx = blockIdx.x * blockDim.x + threadIdx.x;
  if (idx >= 8 * 768) return;
  const int bb = idx / 768, d = idx % 768;
  const float sd = sqrtf(vbuf[idx] + EPSX);
  for (int p = 0; p < 9; ++p) {
    const size_t o = ((size_t)bb * 9 + p) * 768 + d;
    x[o] = (x[o] - b[d]) / g[d] * sd + mbuf[idx];
  }
}

__global__ void k_bcast_symbols(const float* __restrict__ sym, float* __restrict__ y)
{
  const size_t total = (size_t)72 * 768;
  for (size_t idx = (size_t)blockIdx.x * blockDim.x + threadIdx.x; idx < total;
       idx += (size_t)gridDim.x * blockDim.x) {
    const int d = (int)(idx % 768);
    const int p = (int)((idx / 768) % 9);
    y[idx] = sym[(size_t)p * 768 + d];
  }
}

// attention core (B=8, H=32, L=9, dh=24): one block per (b,h)
__global__ __launch_bounds__(32)
void k_attention(const float* __restrict__ q, const float* __restrict__ k,
                 const float* __restrict__ v, float* __restrict__ o,
                 int Bn, int Hh, int L, int dh)
{
  const int b = blockIdx.x / Hh, h = blockIdx.x % Hh;
  const float* qb = q + ((size_t)b * Hh + h) * L * dh;
  const float* kb = k + ((size_t)b * Hh + h) * L * dh;
  const float* vb = v + ((size_t)b * Hh + h) * L * dh;
  __shared__ float s[9][9];
  const int t = threadIdx.x;
  const float scale = rsqrtf((float)dh);
  for (int idx = t; idx < L * L; idx += 32) {
    const int i = idx / L, j = idx % L;
    float acc = 0.f;
    for (int c = 0; c < dh; ++c) acc += qb[i * dh + c] * kb[j * dh + c];
    s[i][j] = acc * scale;
  }
  __syncthreads();
  if (t < L) {
    float mx = -1e30f;
    for (int j = 0; j < L; ++j) mx = fmaxf(mx, s[t][j]);
    float sum = 0.f;
    for (int j = 0; j < L; ++j) { const float e = expf(s[t][j] - mx); s[t][j] = e; sum += e; }
    const float inv = 1.f / sum;
    for (int j = 0; j < L; ++j) s[t][j] *= inv;
  }
  __syncthreads();
  for (int idx = t; idx < L * dh; idx += 32) {
    const int i = idx / dh, c = idx % dh;
    float acc = 0.f;
    for (int j = 0; j < L; ++j) acc += s[i][j] * vb[j * dh + c];
    o[((size_t)b * L + i) * (Hh * dh) + h * dh + c] = acc;
  }
}

// scores[b,k] = emb[b,8,:] . cands[b,k,:]
__global__ __launch_bounds__(32)
void k_scores(const float* __restrict__ emb, const float* __restrict__ cands,
              float* __restrict__ out)
{
  const int b = blockIdx.x / 8, kk = blockIdx.x % 8;
  const float* e = emb + ((size_t)b * 9 + 8) * 768;
  const float* c = cands + ((size_t)b * 8 + kk) * 768;
  float s = 0.f;
  for (int d = threadIdx.x; d < 768; d += 32) s += e[d] * c[d];
  for (int o = 16; o > 0; o >>= 1) s += __shfl_xor(s, o, 32);
  if (threadIdx.x == 0) out[b * 8 + kk] = s;
}

// =====================================================================
// host-side orchestration
// =====================================================================
static inline int up32(int x)  { return (x + 31) & ~31; }
static inline int up64(int x)  { return (x + 63) & ~63; }
static inline int up128(int x) { return (x + 127) & ~127; }

static inline unsigned gsz(size_t n)
{
  size_t b = (n + 255) / 256;
  if (b > 32768) b = 32768;
  if (b == 0) b = 1;
  return (unsigned)b;
}

static void gemm(hipStream_t s, const _Float16* A, const _Float16* B, const float* bias,
                 float* Cf, _Float16* Ch, int M, int N, int Kp, int act)
{
  dim3 grid((unsigned)(up64(N) / 64), (unsigned)(up128(M) / 128));
  if (act == 0)      k_gemm<0><<<grid, 256, 0, s>>>(A, B, bias, Cf, Ch, M, N, Kp);
  else if (act == 1) k_gemm<1><<<grid, 256, 0, s>>>(A, B, bias, Cf, Ch, M, N, Kp);
  else if (act == 2) k_gemm<2><<<grid, 256, 0, s>>>(A, B, bias, Cf, Ch, M, N, Kp);
  else               k_gemm<3><<<grid, 256, 0, s>>>(A, B, bias, Cf, Ch, M, N, Kp);
}

struct Ctx {
  hipStream_t s;
  void* const* in;
  _Float16* wbuf;
  float *bufA, *bufB, *bufC, *bufD;
  _Float16 *t16a, *t16b, *t16c, *t16o, *t16h;
};
static inline const float* F(const Ctx& c, int i) { return (const float*)c.in[i]; }

// transformer block: out = block_fwd(xq, xk, xv) with param base pb
static void run_block(Ctx& c, const float* xq, const float* xk, const float* xv,
                      float* out, int pb)
{
  const int D = 768, Mr = 72;
  k_layernorm<<<Mr, 32, 0, c.s>>>(xq, F(c, pb + 0), F(c, pb + 1), nullptr, c.t16a, D);
  k_layernorm<<<Mr, 32, 0, c.s>>>(xk, F(c, pb + 0), F(c, pb + 1), nullptr, c.t16b, D);
  k_layernorm<<<Mr, 32, 0, c.s>>>(xv, F(c, pb + 2), F(c, pb + 3), nullptr, c.t16c, D);
  // q, k, v projections
  k_wlin<<<gsz((size_t)D * D), 256, 0, c.s>>>(F(c, pb + 8), c.wbuf, D, D, 0);
  gemm(c.s, c.t16a, c.wbuf, F(c, pb + 9), c.bufA, nullptr, Mr, D, D, 0);
  k_wlin<<<gsz((size_t)D * D), 256, 0, c.s>>>(F(c, pb + 10), c.wbuf, D, D, 0);
  gemm(c.s, c.t16b, c.wbuf, F(c, pb + 11), c.bufB, nullptr, Mr, D, D, 0);
  k_wlin<<<gsz((size_t)D * D), 256, 0, c.s>>>(F(c, pb + 12), c.wbuf, D, D, 0);
  gemm(c.s, c.t16c, c.wbuf, F(c, pb + 13), c.bufC, nullptr, Mr, D, D, 0);
  // attention core + output projection
  k_attention<<<8 * 32, 32, 0, c.s>>>(c.bufA, c.bufB, c.bufC, c.bufD, 8, 32, 9, 24);
  k_f32_to_f16<<<gsz((size_t)Mr * D), 256, 0, c.s>>>(c.bufD, c.t16o, (size_t)Mr * D);
  k_wlin<<<gsz((size_t)D * D), 256, 0, c.s>>>(F(c, pb + 14), c.wbuf, D, D, 0);
  gemm(c.s, c.t16o, c.wbuf, F(c, pb + 15), c.bufA, nullptr, Mr, D, D, 0);
  // x = xv + attn
  k_add<<<gsz((size_t)Mr * D), 256, 0, c.s>>>(xv, c.bufA, c.bufB, (size_t)Mr * D);
  // MLP
  k_layernorm<<<Mr, 32, 0, c.s>>>(c.bufB, F(c, pb + 4), F(c, pb + 5), nullptr, c.t16a, D);
  k_wlin<<<gsz((size_t)3072 * D), 256, 0, c.s>>>(F(c, pb + 16), c.wbuf, 3072, D, 0);
  gemm(c.s, c.t16a, c.wbuf, F(c, pb + 17), nullptr, c.t16h, Mr, 3072, D, 2); // gelu -> f16
  k_wlin<<<gsz((size_t)D * 3072), 256, 0, c.s>>>(F(c, pb + 18), c.wbuf, D, 3072, 0);
  gemm(c.s, c.t16h, c.wbuf, F(c, pb + 19), c.bufC, nullptr, Mr, D, 3072, 0);
  k_add<<<gsz((size_t)Mr * D), 256, 0, c.s>>>(c.bufB, c.bufC, c.bufD, (size_t)Mr * D);
  k_layernorm<<<Mr, 32, 0, c.s>>>(c.bufD, F(c, pb + 6), F(c, pb + 7), out, nullptr, D);
}

extern "C" void kernel_launch(void* const* d_in, const int* in_sizes, int n_in,
                              void* d_out, int out_size, void* d_ws, size_t ws_size,
                              hipStream_t stream)
{
  (void)in_sizes; (void)n_in; (void)out_size; (void)ws_size;
  char* ws = (char*)d_ws;
  size_t off = 0;
  auto alloc = [&](size_t bytes) -> char* {
    char* p = ws + off;
    off += (bytes + 255) & ~(size_t)255;
    return p;
  };

  _Float16* colbuf = (_Float16*)alloc((size_t)96 << 20);
  _Float16* wbuf   = (_Float16*)alloc((size_t)64 << 20);
  float* g0 = (float*)alloc((size_t)256 << 20);
  float* g1 = (float*)alloc((size_t)256 << 20);
  _Float16* act0 = (_Float16*)alloc((size_t)128 << 20);
  _Float16* act1 = (_Float16*)alloc((size_t)128 << 20);
  float* stats = (float*)alloc((size_t)16 << 10);
  // small transformer/decoder scratch
  float* emb   = (float*)alloc((size_t)136 * 768 * 4);
  float* xalt  = (float*)alloc((size_t)72 * 768 * 4);
  float* y0    = (float*)alloc((size_t)72 * 768 * 4);
  float* y1    = (float*)alloc((size_t)72 * 768 * 4);
  float* bufA  = (float*)alloc((size_t)72 * 768 * 4);
  float* bufB  = (float*)alloc((size_t)72 * 768 * 4);
  float* bufC  = (float*)alloc((size_t)72 * 768 * 4);
  float* bufD  = (float*)alloc((size_t)72 * 768 * 4);
  float* yfin  = (float*)alloc((size_t)72 * 768 * 4);
  float* xfin  = (float*)alloc((size_t)72 * 768 * 4);
  float* xsum  = (float*)alloc((size_t)72 * 768 * 4);
  float* embO  = (float*)alloc((size_t)72 * 768 * 4);
  float* tm    = (float*)alloc((size_t)8 * 768 * 4);
  float* tv    = (float*)alloc((size_t)8 * 768 * 4);
  float* biasP = (float*)alloc((size_t)25600 * 4);
  // padded f16 A-side buffers (Mp = 128 rows; up128(72) == 128)
  _Float16* t16a = (_Float16*)alloc((size_t)128 * 768 * 2);
  _Float16* t16b = (_Float16*)alloc((size_t)128 * 768 * 2);
  _Float16* t16c = (_Float16*)alloc((size_t)128 * 768 * 2);
  _Float16* t16o = (_Float16*)alloc((size_t)128 * 768 * 2);
  _Float16* t16h = (_Float16*)alloc((size_t)128 * 3072 * 2);
  _Float16* x16  = (_Float16*)alloc((size_t)128 * 768 * 2);
  _Float16* e16  = (_Float16*)alloc((size_t)128 * 768 * 2);

  Ctx c{stream, d_in, wbuf, bufA, bufB, bufC, bufD, t16a, t16b, t16c, t16o, t16h};
  hipStream_t s = stream;

  // zero the padded f16 buffers once: producers only write rows < 72,
  // so the pad rows stay zero for every GEMM that consumes them.
  k_zero_f16<<<gsz((size_t)128 * 768), 256, 0, s>>>(t16a, (size_t)128 * 768);
  k_zero_f16<<<gsz((size_t)128 * 768), 256, 0, s>>>(t16b, (size_t)128 * 768);
  k_zero_f16<<<gsz((size_t)128 * 768), 256, 0, s>>>(t16c, (size_t)128 * 768);
  k_zero_f16<<<gsz((size_t)128 * 768), 256, 0, s>>>(t16o, (size_t)128 * 768);
  k_zero_f16<<<gsz((size_t)128 * 3072), 256, 0, s>>>(t16h, (size_t)128 * 3072);
  k_zero_f16<<<gsz((size_t)128 * 768), 256, 0, s>>>(x16, (size_t)128 * 768);
  k_zero_f16<<<gsz((size_t)128 * 768), 256, 0, s>>>(e16, (size_t)128 * 768);

  // ---------------- encoder: 136 images of (1,160,160) ----------------
  k_f32_to_f16<<<gsz((size_t)72 * 25600), 256, 0, s>>>((const float*)d_in[0], act0,
                                                       (size_t)72 * 25600);
  k_f32_to_f16<<<gsz((size_t)64 * 25600), 256, 0, s>>>((const float*)d_in[1],
                                                       act0 + (size_t)72 * 25600,
                                                       (size_t)64 * 25600);
  const int ci_[5] = {1, 16, 32, 64, 128};
  const int co_[5] = {16, 32, 64, 128, 256};
  const int st_[5] = {1, 2, 2, 2, 2};
  int Hin = 160;
  for (int blk = 0; blk < 5; ++blk) {
    const int base = 2 + blk * 9;
    const int Ci = ci_[blk], Co = co_[blk], st = st_[blk];
    const int Ho = Hin / st;
    const size_t Mtot = (size_t)136 * Ho * Ho;
    const int Mc = 8 * Ho * Ho;
    const int Mp = up128(Mc);
    const int Np = up64(Co);
    const int Kp1 = up32(9 * Ci);
    // conv1 (3x3, stride st, pad 1)
    k_wconv<<<gsz((size_t)Np * Kp1), 256, 0, s>>>(F(c, base + 0), wbuf, Co, Ci, 3, 3, Kp1, Np);
    for (int ch = 0; ch < 17; ++ch) {
      const _Float16* xin = act0 + (size_t)ch * 8 * Hin * Hin * Ci;
      k_im2col<<<gsz((size_t)Mp * Kp1), 256, 0, s>>>(xin, colbuf, 8, Hin, Hin, Ci,
                                                     Ho, Ho, 3, 3, st, 1, Kp1, Mp);
      gemm(s, colbuf, wbuf, nullptr, g0 + (size_t)ch * Mc * Co, nullptr, Mc, Co, Kp1, 0);
    }
    k_zero<<<gsz(2 * Co), 256, 0, s>>>(stats, 2 * Co);
    k_bnstats<<<dim3(64, Co), 256, 0, s>>>(g0, stats, stats + Co, (long long)Mtot, Co);
    k_bn_apply_relu<<<gsz(Mtot * Co), 256, 0, s>>>(g0, stats, stats + Co,
                                                   F(c, base + 1), F(c, base + 2),
                                                   act1, Mtot, Co);
    // conv2 (3x3, stride 1, pad 1) on act1
    const int Kp2 = up32(9 * Co);
    k_wconv<<<gsz((size_t)Np * Kp2), 256, 0, s>>>(F(c, base + 3), wbuf, Co, Co, 3, 3, Kp2, Np);
    for (int ch = 0; ch < 17; ++ch) {
      const _Float16* xin = act1 + (size_t)ch * 8 * Ho * Ho * Co;
      k_im2col<<<gsz((size_t)Mp * Kp2), 256, 0, s>>>(xin, colbuf, 8, Ho, Ho, Co,
                                                     Ho, Ho, 3, 3, 1, 1, Kp2, Mp);
      gemm(s, colbuf, wbuf, nullptr, g0 + (size_t)ch * Mc * Co, nullptr, Mc, Co, Kp2, 0);
    }
    // shortcut (1x1, stride st, pad 0) on act0
    const int Kps = up32(Ci);
    k_wconv<<<gsz((size_t)Np * Kps), 256, 0, s>>>(F(c, base + 6), wbuf, Co, Ci, 1, 1, Kps, Np);
    for (int ch = 0; ch < 17; ++ch) {
      const _Float16* xin = act0 + (size_t)ch * 8 * Hin * Hin * Ci;
      k_im2col<<<gsz((size_t)Mp * Kps), 256, 0, s>>>(xin, colbuf, 8, Hin, Hin, Ci,
                                                     Ho, Ho, 1, 1, st, 0, Kps, Mp);
      gemm(s, colbuf, wbuf, nullptr, g1 + (size_t)ch * Mc * Co, nullptr, Mc, Co, Kps, 0);
    }
    // bn2 + bn_sc + add + relu -> act0 (f16), becomes next block input
    k_zero<<<gsz(4 * Co), 256, 0, s>>>(stats, 4 * Co);
    k_bnstats<<<dim3(64, Co), 256, 0, s>>>(g0, stats, stats + Co, (long long)Mtot, Co);
    k_bnstats<<<dim3(64, Co), 256, 0, s>>>(g1, stats + 2 * Co, stats + 3 * Co, (long long)Mtot, Co);
    k_bn2_add_relu<<<gsz(Mtot * Co), 256, 0, s>>>(g0, g1,
        stats, stats + Co, F(c, base + 4), F(c, base + 5),
        stats + 2 * Co, stats + 3 * Co, F(c, base + 7), F(c, base + 8),
        act0, Mtot, Co);
    Hin = Ho;
  }
  // encoder fc: (136, 25600) -> (136, 768); zero-pad rows 136..255 of A
  k_zero_f16<<<gsz((size_t)120 * 25600), 256, 0, s>>>(act0 + (size_t)136 * 25600,
                                                      (size_t)120 * 25600);
  k_wlin<<<gsz((size_t)768 * 25600), 256, 0, s>>>(F(c, 47), wbuf, 768, 25600, 1);
  gemm(s, act0, wbuf, F(c, 48), emb, nullptr, 136, 768, 25600, 0);

  // ---------------- transformer ----------------
  float* x = emb;                           // rows 0..71 = (8,9,768)
  float* cands_e = emb + (size_t)72 * 768;  // rows 72..135 = (8,8,768)
  k_posembed<<<gsz((size_t)72 * 768), 256, 0, s>>>(x);
  k_tcn_fwd<<<gsz(8 * 768), 256, 0, s>>>(x, tm, tv, F(c, 49), F(c, 50));
  k_bcast_symbols<<<gsz((size_t)72 * 768), 256, 0, s>>>(F(c, 51), y0);
  run_block(c, x, x, y0, y1, 52);           // relBottleneck
  float* yc = y1; float* ya = y0;
  for (int i = 0; i < 3; ++i) { run_block(c, yc, yc, yc, ya, 72 + 20 * i); float* t = yc; yc = ya; ya = t; }
  float* xc = x; float* xa = xalt;
  for (int i = 0; i < 4; ++i) { run_block(c, xc, xc, xc, xa, 132 + 20 * i); float* t = xc; xc = xa; xa = t; }
  k_layernorm<<<72, 32, 0, s>>>(yc, F(c, 212), F(c, 213), yfin, nullptr, 768); // norm_x on y
  k_layernorm<<<72, 32, 0, s>>>(xc, F(c, 214), F(c, 215), xfin, nullptr, 768); // norm_y on x
  k_add<<<gsz((size_t)72 * 768), 256, 0, s>>>(xfin, yfin, xsum, (size_t)72 * 768);
  k_tcn_inv<<<gsz(8 * 768), 256, 0, s>>>(xsum, tm, tv, F(c, 49), F(c, 50));
  k_f32_to_f16<<<gsz((size_t)72 * 768), 256, 0, s>>>(xsum, x16, (size_t)72 * 768);
  k_wlin<<<gsz((size_t)768 * 768), 256, 0, s>>>(F(c, 216), wbuf, 768, 768, 0);
  gemm(s, x16, wbuf, F(c, 217), embO, nullptr, 72, 768, 768, 0);

  // scores
  k_scores<<<64, 32, 0, s>>>(embO, cands_e, (float*)d_out + (size_t)72 * 160 * 160);

  // ---------------- decoder ----------------
  k_f32_to_f16<<<gsz((size_t)72 * 768), 256, 0, s>>>(embO, e16, (size_t)72 * 768);
  k_wlin<<<gsz((size_t)25600 * 768), 256, 0, s>>>(F(c, 218), wbuf, 25600, 768, 2);
  k_decbias<<<gsz(25600), 256, 0, s>>>(F(c, 219), biasP);
  gemm(s, e16, wbuf, biasP, nullptr, act0, 72, 25600, 768, 0); // (72,10,10,256) f16
  const int dci[4] = {256, 128, 64, 32};
  const int dco[4] = {128, 64, 32, 16};
  int H = 10;
  _Float16* din = act0; _Float16* dout = act1;
  for (int l = 0; l < 4; ++l) {
    const int Ci = dci[l], Co = dco[l], Ho = 2 * H;
    const int Kd = 9 * Ci;                  // multiple of 32
    const int Np = up64(Co);
    k_wdeconv<<<gsz((size_t)Np * Kd), 256, 0, s>>>(F(c, 220 + 2 * l), wbuf, Ci, Co, Np);
    for (int ch = 0; ch < 18; ++ch) {
      const _Float16* xin = din + (size_t)ch * 4 * H * H * Ci;
      const int Mc = 4 * Ho * Ho;
      const int Mp = up128(Mc);
      k_im2col_dec<<<gsz((size_t)Mp * Kd), 256, 0, s>>>(xin, colbuf, 4, H, H, Ci, Ho, Ho, Mp);
      gemm(s, colbuf, wbuf, F(c, 221 + 2 * l), nullptr, dout + (size_t)ch * Mc * Co,
           Mc, Co, Kd, 1);
    }
    { _Float16* t = din; din = dout; dout = t; }
    H = Ho;
  }
  // final conv (1,16,3,3) + sigmoid -> rec in d_out
  const int Kpf = up32(144);
  k_wconv<<<gsz((size_t)64 * Kpf), 256, 0, s>>>(F(c, 228), wbuf, 1, 16, 3, 3, Kpf, 64);
  for (int ch = 0; ch < 18; ++ch) {
    const _Float16* xin = din + (size_t)ch * 4 * 160 * 160 * 16;
    const int Mc = 4 * 160 * 160;
    k_im2col<<<gsz((size_t)Mc * Kpf), 256, 0, s>>>(xin, colbuf, 4, 160, 160, 16,
                                                   160, 160, 3, 3, 1, 1, Kpf, Mc);
    gemm(s, colbuf, wbuf, F(c, 229), (float*)d_out + (size_t)ch * Mc, nullptr,
         Mc, 1, Kpf, 3);
  }
}
